// InputPairStackBlock_71201967833279
// MI455X (gfx1250) — compile-verified, matrix operand order checked by reading, not run on hardware
//
#include <hip/hip_runtime.h>
#include <cstdint>

// ---------------------------------------------------------------------------
// AlphaFold pair-stack block (tri-att start/end, tri-mul out/in, transition)
// for MI455X (gfx1250, wave32, WMMA).
//
// Roofline: ~140 GFLOP of GEMMs vs ~1.5 GB HBM traffic -> 23.3 TB/s gives a
// ~50us memory floor; bf16 WMMA (v_wmma_f32_16x16x32_bf16, f32 accumulate)
// puts compute in the same range. Every matmul runs through one batched WMMA
// GEMM kernel: bf16 tiles staged in LDS (async-to-LDS when the toolchain
// exposes the gfx1250 builtins, else b128 load/store), fragments read as
// contiguous ds_load_b128 pairs, fused f32 epilogue
// (bias/relu/sigmoid/scale/row-mask/residual-add). LN/softmax/gates stay fp32.
// ---------------------------------------------------------------------------

#define NRES   256
#define NN     (NRES * NRES)     // 65536 pair positions
#define CT     128               // pair channel
#define NHEADS 4
#define CATT   32
#define CHUNK  32                // attention i-slab (8 slabs)

typedef __attribute__((ext_vector_type(16))) __bf16 v16bf;
typedef __attribute__((ext_vector_type(8)))  float  v8f;

#if defined(__gfx1250__) && __has_builtin(__builtin_amdgcn_global_load_async_to_lds_b128) && \
    __has_builtin(__builtin_amdgcn_global_load_async_to_lds_b64)
#define USE_ASYNC_LDS 1
#else
#define USE_ASYNC_LDS 0
#endif

#if USE_ASYNC_LDS
// builtin signatures (from clang diagnostics):
//   b128: (addrspace(1) int __vector_size__(16)*, addrspace(3) int __vector_size__(16)*, int, int)
//   b64 : (addrspace(1) int __vector_size__(8)*,  addrspace(3) int __vector_size__(8)*,  int, int)
typedef int a_v4i __attribute__((vector_size(16)));
typedef int a_v2i __attribute__((vector_size(8)));
typedef __attribute__((address_space(1))) a_v4i* gptr128_t;
typedef __attribute__((address_space(1))) a_v2i* gptr64_t;
typedef __attribute__((address_space(3))) a_v4i* lptr128_t;
typedef __attribute__((address_space(3))) a_v2i* lptr64_t;

__device__ inline gptr128_t glob128(const void* p) { return (gptr128_t)(uintptr_t)p; }
__device__ inline gptr64_t  glob64(const void* p)  { return (gptr64_t)(uintptr_t)p; }
// generic pointers to LDS carry the LDS offset in the low 32 bits
__device__ inline lptr128_t lds128(void* p) { return (lptr128_t)(uint32_t)(uintptr_t)p; }
__device__ inline lptr64_t  lds64(void* p)  { return (lptr64_t)(uint32_t)(uintptr_t)p; }
#endif

// ------------------------------- elementwise -------------------------------

__global__ void f32_to_bf16_kernel(const float* __restrict__ s, __bf16* __restrict__ d, long n) {
  long i  = (long)blockIdx.x * blockDim.x + threadIdx.x;
  long st = (long)gridDim.x * blockDim.x;
  for (; i < n; i += st) d[i] = (__bf16)s[i];
}

// weights: src [K,N] f32 -> dst [N,K] bf16 (so all weight GEMMs take the
// vectorized NT staging path)
__global__ void f32_to_bf16_t_kernel(const float* __restrict__ s, __bf16* __restrict__ d,
                                     int K, int N) {
  long tot = (long)K * N;
  for (long idx = (long)blockIdx.x * blockDim.x + threadIdx.x; idx < tot;
       idx += (long)gridDim.x * blockDim.x) {
    int k = (int)(idx % K);
    int n = (int)(idx / K);
    d[idx] = (__bf16)s[(long)k * N + n];  // idx == n*K + k
  }
}

__global__ void transpose_pair_kernel(const float* __restrict__ s, float* __restrict__ d) {
  // d[j,i,c] = s[i,j,c]
  long tot = (long)NN * CT;
  for (long idx = (long)blockIdx.x * blockDim.x + threadIdx.x; idx < tot;
       idx += (long)gridDim.x * blockDim.x) {
    int  c = (int)(idx & (CT - 1));
    long p = idx >> 7;
    int  j = (int)(p & (NRES - 1));
    int  i = (int)(p >> 8);
    d[(((long)j * NRES) + i) * CT + c] = s[idx];
  }
}

__global__ void transpose_add_pair_kernel(const float* __restrict__ sT, float* __restrict__ d) {
  // d[i,j,c] += sT[j,i,c]
  long tot = (long)NN * CT;
  for (long idx = (long)blockIdx.x * blockDim.x + threadIdx.x; idx < tot;
       idx += (long)gridDim.x * blockDim.x) {
    int  c = (int)(idx & (CT - 1));
    long p = idx >> 7;
    int  j = (int)(p & (NRES - 1));
    int  i = (int)(p >> 8);
    d[idx] += sT[(((long)j * NRES) + i) * CT + c];
  }
}

__global__ void transpose_mask_kernel(const float* __restrict__ s, float* __restrict__ d) {
  int idx = blockIdx.x * blockDim.x + threadIdx.x;
  if (idx < NN) {
    int j = idx & (NRES - 1);
    int i = idx >> 8;
    d[(long)j * NRES + i] = s[idx];
  }
}

__global__ void gate_mul_pack_kernel(const float* __restrict__ o, const float* __restrict__ g,
                                     __bf16* __restrict__ out, long n) {
  long i  = (long)blockIdx.x * blockDim.x + threadIdx.x;
  long st = (long)gridDim.x * blockDim.x;
  for (; i < n; i += st) out[i] = (__bf16)(o[i] * g[i]);
}

__global__ void mul_add_residual_kernel(float* __restrict__ z, const float* __restrict__ a,
                                        const float* __restrict__ b, long n) {
  long i  = (long)blockIdx.x * blockDim.x + threadIdx.x;
  long st = (long)gridDim.x * blockDim.x;
  for (; i < n; i += st) z[i] += a[i] * b[i];
}

// a = mask * p * q  written transposed+packed to bf16 [c, r, t]
// srcT==0: source position (r,t); srcT==1: source position (t,r)
__global__ void combine_pack_t_kernel(const float* __restrict__ p, const float* __restrict__ q,
                                      const float* __restrict__ m, __bf16* __restrict__ out,
                                      int srcT) {
  long tot = (long)NN * CT;
  for (long idx = (long)blockIdx.x * blockDim.x + threadIdx.x; idx < tot;
       idx += (long)gridDim.x * blockDim.x) {
    int  t    = (int)(idx & (NRES - 1));
    long rest = idx >> 8;
    int  r    = (int)(rest & (NRES - 1));
    int  c    = (int)(rest >> 8);
    long pos  = srcT ? ((long)t * NRES + r) : ((long)r * NRES + t);
    float v   = m[pos] * p[pos * CT + c] * q[pos * CT + c];
    out[idx]  = (__bf16)v;  // idx == (c*NRES + r)*NRES + t  -> [c][r][t]
  }
}

// ------------------------------ LayerNorm ----------------------------------
// One wave per pair position (32 lanes x 4 channels), __shfl_xor reductions.

__global__ __launch_bounds__(256)
void layernorm_kernel(const float* __restrict__ x, const float* __restrict__ g,
                      const float* __restrict__ b, __bf16* __restrict__ out) {
  int  lane = threadIdx.x & 31;
  long pos  = (long)blockIdx.x * 8 + (threadIdx.x >> 5);
  const float* xp = x + pos * CT;
  float4 v = *(const float4*)(xp + lane * 4);
  float  s = v.x + v.y + v.z + v.w;
#pragma unroll
  for (int o = 16; o > 0; o >>= 1) s += __shfl_xor(s, o, 32);
  float mean = s * (1.0f / CT);
  float dx = v.x - mean, dy = v.y - mean, dz = v.z - mean, dw = v.w - mean;
  float vs = dx * dx + dy * dy + dz * dz + dw * dw;
#pragma unroll
  for (int o = 16; o > 0; o >>= 1) vs += __shfl_xor(vs, o, 32);
  float rs = rsqrtf(vs * (1.0f / CT) + 1e-5f);
  float4 gg = *(const float4*)(g + lane * 4);
  float4 bb = *(const float4*)(b + lane * 4);
  __bf16* op = out + pos * CT + lane * 4;
  op[0] = (__bf16)(dx * rs * gg.x + bb.x);
  op[1] = (__bf16)(dy * rs * gg.y + bb.y);
  op[2] = (__bf16)(dz * rs * gg.z + bb.z);
  op[3] = (__bf16)(dw * rs * gg.w + bb.w);
}

// LayerNorm over c of xT[c, pos] (tri-mul output in [c,i,j]), emit bf16 [pos,c]
__global__ __launch_bounds__(256)
void ln_perm_kernel(const float* __restrict__ xT, const float* __restrict__ g,
                    const float* __restrict__ b, __bf16* __restrict__ out) {
  int  lane = threadIdx.x & 31;
  long pos  = (long)blockIdx.x * 8 + (threadIdx.x >> 5);
  float v[4]; float s = 0.f;
#pragma unroll
  for (int j = 0; j < 4; ++j) { v[j] = xT[(long)(lane * 4 + j) * NN + pos]; s += v[j]; }
#pragma unroll
  for (int o = 16; o > 0; o >>= 1) s += __shfl_xor(s, o, 32);
  float mean = s * (1.0f / CT);
  float vs = 0.f;
#pragma unroll
  for (int j = 0; j < 4; ++j) { float d = v[j] - mean; vs += d * d; }
#pragma unroll
  for (int o = 16; o > 0; o >>= 1) vs += __shfl_xor(vs, o, 32);
  float rs = rsqrtf(vs * (1.0f / CT) + 1e-5f);
#pragma unroll
  for (int j = 0; j < 4; ++j)
    out[pos * CT + lane * 4 + j] =
        (__bf16)((v[j] - mean) * rs * g[lane * 4 + j] + b[lane * 4 + j]);
}

// tb[h, q*256+k] = dot(zl[q,k,:], wb[:,h])
__global__ void tri_bias_kernel(const __bf16* __restrict__ zl, const float* __restrict__ wb,
                                float* __restrict__ tb) {
  long m = (long)blockIdx.x * blockDim.x + threadIdx.x;
  if (m >= NN) return;
  float a0 = 0.f, a1 = 0.f, a2 = 0.f, a3 = 0.f;
  const __bf16* zp = zl + m * CT;
  for (int c = 0; c < CT; ++c) {
    float zv = (float)zp[c];
    const float* w = wb + c * NHEADS;
    a0 += zv * w[0]; a1 += zv * w[1]; a2 += zv * w[2]; a3 += zv * w[3];
  }
  tb[m] = a0; tb[(long)NN + m] = a1; tb[2L * NN + m] = a2; tb[3L * NN + m] = a3;
}

// softmax over k with mask bias + triangle bias; one block per (ii,h,q) row
__global__ __launch_bounds__(256)
void softmax_bias_kernel(const float* __restrict__ s, const float* __restrict__ tb,
                         const float* __restrict__ maskp, __bf16* __restrict__ out, int i0) {
  long ri = blockIdx.x;
  int  q  = (int)(ri & (NRES - 1));
  int  h  = (int)((ri >> 8) & (NHEADS - 1));
  int  ii = (int)(ri >> 10);
  int  i  = i0 + ii;
  int  t  = threadIdx.x;
  float v = s[ri * NRES + t] + tb[(long)h * NN + (long)q * NRES + t] +
            1e9f * (maskp[(long)i * NRES + t] - 1.0f);
  __shared__ float red[256];
  red[t] = v; __syncthreads();
#pragma unroll
  for (int o = 128; o > 0; o >>= 1) { if (t < o) red[t] = fmaxf(red[t], red[t + o]); __syncthreads(); }
  float mx = red[0]; __syncthreads();
  float e = __expf(v - mx);
  red[t] = e; __syncthreads();
#pragma unroll
  for (int o = 128; o > 0; o >>= 1) { if (t < o) red[t] += red[t + o]; __syncthreads(); }
  out[ri * NRES + t] = (__bf16)(e / red[0]);
}

// ------------------------- generic batched WMMA GEMM -----------------------
// C[b] = act(scale * A[b](MxK) @ B[b](KxN or NxK^T) + bias) [* maskRow] [+= outF]
// Block: 256 threads = 8 waves; tile 128(M) x 32(N) x 32(K); 2 WMMA per wave/step.
// LDS layouts chosen so every fragment is two contiguous 16B runs per lane:
//   As[m][k]  (row-major)  -> A frag = 2x ds_load_b128
//   Bs[n][k]  (transposed) -> B frag = 2x ds_load_b128
// Staging uses GLOBAL_LOAD_ASYNC_TO_LDS (ASYNCcnt) when the builtins exist.

#define BM 128
#define BN 32
#define BK 32
#define LPAD 40  // padded LDS row (halves): 80B = 5 x 16B, keeps b128 alignment

__device__ inline v16bf load_frag(const __bf16* base) {
  union { uint4 u[2]; v16bf v; } t;
  t.u[0] = *(const uint4*)(base);       // halves [0..7]   (k = kb .. kb+7)
  t.u[1] = *(const uint4*)(base + 16);  // halves [8..15]  (k = kb+16 .. kb+23)
  return t.v;
}

template <int TRANSB>
__global__ __launch_bounds__(256)
void wmma_gemm_kernel(const __bf16* __restrict__ A, long aBase, int lda, long aSI, long aSH,
                      const __bf16* __restrict__ B, long bBase, int ldb, long bSI, long bSH,
                      float* outF, __bf16* outH, long oBase, int ldo, long oSI, long oSH,
                      int addTo,
                      const float* __restrict__ bias, const float* __restrict__ maskRow,
                      long maskBase, int act, float scale,
                      int M, int N, int K, int nh) {
  __shared__ __align__(16) __bf16 As[BM][LPAD];
  __shared__ __align__(16) __bf16 Bs[BN][LPAD];

  int  bz = blockIdx.z;
  int  h  = bz % nh;
  int  bi = bz / nh;
  long aOff = aBase + (long)bi * aSI + (long)h * aSH;
  long bOff = bBase + (long)bi * bSI + (long)h * bSH;
  long oOff = oBase + (long)bi * oSI + (long)h * oSH;

  int tid  = threadIdx.x;
  int wave = tid >> 5, lane = tid & 31;
  int m0 = blockIdx.y * BM, n0 = blockIdx.x * BN;

  v8f acc0 = {0.f, 0.f, 0.f, 0.f, 0.f, 0.f, 0.f, 0.f};
  v8f acc1 = acc0;

  int arow = tid >> 1;         // 128 rows, 2 threads/row
  int acol = (tid & 1) * 16;
  int brow = tid >> 3;         // 32 rows, 8 threads/row
  int bcol = (tid & 7) * 4;

  // fragment addressing (ISA 16-bit 16x32 layout):
  // lanes 0-15: k = 0..7 / 16..23 ; lanes 16-31: k = 8..15 / 24..31
  int mrow = wave * 16 + (lane & 15);
  int kb   = (lane >> 4) * 8;
  int ncol = lane & 15;

  for (int k0 = 0; k0 < K; k0 += BK) {
    const __bf16* gpa = A + aOff + (long)(m0 + arow) * lda + k0 + acol;
#if USE_ASYNC_LDS
    // async DMA into LDS, tracked by ASYNCcnt (no VGPR round-trip)
    __builtin_amdgcn_global_load_async_to_lds_b128(glob128(gpa),
                                                   lds128(&As[arow][acol]), 0, 0);
    __builtin_amdgcn_global_load_async_to_lds_b128(glob128(gpa + 8),
                                                   lds128(&As[arow][acol + 8]), 0, 0);
#else
    {
      const uint4* ap = (const uint4*)gpa;
      *(uint4*)&As[arow][acol]     = ap[0];
      *(uint4*)&As[arow][acol + 8] = ap[1];
    }
#endif
    __builtin_prefetch(gpa + BK, 0, 1);
    // stage B tile into Bs[n][k]
    if (TRANSB) {
      // B is [N,K]: straight vectorized copy
      const __bf16* gpb = B + bOff + (long)(n0 + brow) * ldb + k0 + bcol;
#if USE_ASYNC_LDS
      __builtin_amdgcn_global_load_async_to_lds_b64(glob64(gpb),
                                                    lds64(&Bs[brow][bcol]), 0, 0);
#else
      *(uint2*)&Bs[brow][bcol] = *(const uint2*)gpb;
#endif
    } else {
      // B is [K,N]: transpose during staging
      const uint2* bp = (const uint2*)(B + bOff + (long)(k0 + brow) * ldb + n0 + bcol);
      uint2 tv = bp[0];
      const __bf16* te = (const __bf16*)&tv;
      Bs[bcol + 0][brow] = te[0];
      Bs[bcol + 1][brow] = te[1];
      Bs[bcol + 2][brow] = te[2];
      Bs[bcol + 3][brow] = te[3];
    }
#if USE_ASYNC_LDS
#if __has_builtin(__builtin_amdgcn_s_wait_asynccnt)
    __builtin_amdgcn_s_wait_asynccnt(0);
#else
    asm volatile("s_wait_asynccnt 0" ::: "memory");
#endif
#endif
    __syncthreads();

    v16bf fa  = load_frag(&As[mrow][kb]);
    v16bf fb0 = load_frag(&Bs[ncol][kb]);
    v16bf fb1 = load_frag(&Bs[ncol + 16][kb]);
    acc0 = __builtin_amdgcn_wmma_f32_16x16x32_bf16(false, fa, false, fb0, (short)0, acc0,
                                                   false, false);
    acc1 = __builtin_amdgcn_wmma_f32_16x16x32_bf16(false, fa, false, fb1, (short)0, acc1,
                                                   false, false);
    __syncthreads();
  }

  // epilogue: C layout — VGPR r: lanes 0-15 => M=r, N=lane; lanes 16-31 => M=8+r, N=lane-16
  int mb = m0 + wave * 16 + ((lane >> 4) * 8);
  int nb = n0 + (lane & 15);
#pragma unroll
  for (int r = 0; r < 8; ++r) {
    int gm = mb + r;
#pragma unroll
    for (int nt = 0; nt < 2; ++nt) {
      float val = (nt ? acc1[r] : acc0[r]) * scale;
      int gn = nb + nt * 16;
      if (bias) val += bias[gn];
      if (act == 1) val = fmaxf(val, 0.f);
      else if (act == 2) val = 1.f / (1.f + __expf(-val));
      if (maskRow) val *= maskRow[maskBase + gm];
      long idx = oOff + (long)gm * ldo + gn;
      if (outF) outF[idx] = addTo ? (outF[idx] + val) : val;
      if (outH) outH[idx] = (__bf16)val;
    }
  }
}

// ------------------------------- host side ---------------------------------

struct WS {
  __bf16 *H0, *H1, *H2, *H3, *H4, *HA;
  float  *F0, *F1, *F2, *F3, *tb, *maskT;
};

static void gemm(hipStream_t st,
                 const __bf16* A, long aBase, int lda, long aSI, long aSH,
                 const __bf16* B, long bBase, int ldb, long bSI, long bSH, int transB,
                 float* outF, __bf16* outH, long oBase, int ldo, long oSI, long oSH, int addTo,
                 const float* bias, const float* maskRow, long maskBase,
                 int act, float scale, int M, int N, int K, int nh, int nI) {
  dim3 grid(N / BN, M / BM, nh * nI);
  if (transB)
    wmma_gemm_kernel<1><<<grid, 256, 0, st>>>(A, aBase, lda, aSI, aSH, B, bBase, ldb, bSI,
                                              bSH, outF, outH, oBase, ldo, oSI, oSH, addTo,
                                              bias, maskRow, maskBase, act, scale, M, N, K, nh);
  else
    wmma_gemm_kernel<0><<<grid, 256, 0, st>>>(A, aBase, lda, aSI, aSH, B, bBase, ldb, bSI,
                                              bSH, outF, outH, oBase, ldo, oSI, oSH, addTo,
                                              bias, maskRow, maskBase, act, scale, M, N, K, nh);
}

struct AttW {
  const float *ln_g, *ln_b, *wb, *bg, *bo;
  const __bf16 *wq, *wk, *wv, *wg, *wo;  // stored transposed [N,K]
};

static void run_attention(const float* src, const float* maskp, const AttW& w, const WS& ws,
                          float* woOut, int addTo, hipStream_t st) {
  layernorm_kernel<<<NN / 8, 256, 0, st>>>(src, w.ln_g, w.ln_b, ws.H0);
  tri_bias_kernel<<<NN / 256, 256, 0, st>>>(ws.H0, w.wb, ws.tb);
  const float qscale = 0.17677669529663687f;  // C_ATT^-0.5
  // q/k/v/gate projections over all 65536 rows (weights transposed -> NT path)
  gemm(st, ws.H0, 0, CT, 0, 0, w.wq, 0, CT, 0, 0, 1, nullptr, ws.H1, 0, CT, 0, 0, 0,
       nullptr, nullptr, 0, 0, qscale, NN, CT, CT, 1, 1);
  gemm(st, ws.H0, 0, CT, 0, 0, w.wk, 0, CT, 0, 0, 1, nullptr, ws.H2, 0, CT, 0, 0, 0,
       nullptr, nullptr, 0, 0, 1.f, NN, CT, CT, 1, 1);
  gemm(st, ws.H0, 0, CT, 0, 0, w.wv, 0, CT, 0, 0, 1, nullptr, ws.H3, 0, CT, 0, 0, 0,
       nullptr, nullptr, 0, 0, 1.f, NN, CT, CT, 1, 1);
  gemm(st, ws.H0, 0, CT, 0, 0, w.wg, 0, CT, 0, 0, 1, ws.F2, nullptr, 0, CT, 0, 0, 0,
       w.bg, nullptr, 0, 2, 1.f, NN, CT, CT, 1, 1);
  // attention core, chunked over i-slabs
  for (int cc = 0; cc < NRES / CHUNK; ++cc) {
    long i0 = (long)cc * CHUNK;
    // scores[ii,h,q,k] = q @ k^T  (batched NT, K=32 -> single WMMA step)
    gemm(st, ws.H1, i0 * NRES * CT, CT, (long)NRES * CT, CATT,
             ws.H2, i0 * NRES * CT, CT, (long)NRES * CT, CATT, 1,
             ws.F3, nullptr, 0, NRES, (long)NHEADS * NRES * NRES, (long)NRES * NRES, 0,
             nullptr, nullptr, 0, 0, 1.f, NRES, NRES, CATT, NHEADS, CHUNK);
    softmax_bias_kernel<<<CHUNK * NHEADS * NRES, 256, 0, st>>>(ws.F3, ws.tb, maskp, ws.HA,
                                                               (int)i0);
    // o[ii,q,h*32+c] = a @ v  (batched NN: B = v activations, scatter staging)
    gemm(st, ws.HA, 0, NRES, (long)NHEADS * NRES * NRES, (long)NRES * NRES,
             ws.H3, i0 * NRES * CT, CT, (long)NRES * CT, CATT, 0,
             ws.F0, nullptr, i0 * NRES * CT, CT, (long)NRES * CT, CATT, 0,
             nullptr, nullptr, 0, 0, 1.f, NRES, CATT, NRES, NHEADS, CHUNK);
  }
  gate_mul_pack_kernel<<<4096, 256, 0, st>>>(ws.F0, ws.F2, ws.H4, (long)NN * CT);
  gemm(st, ws.H4, 0, CT, 0, 0, w.wo, 0, CT, 0, 0, 1, woOut, nullptr, 0, CT, 0, 0, addTo,
       w.bo, nullptr, 0, 0, 1.f, NN, CT, CT, 1, 1);
}

struct MulW {
  const float *ln_in_g, *ln_in_b, *bap, *bag, *bbp, *bbg, *ln_out_g, *ln_out_b, *bz, *bg;
  const __bf16 *wap, *wag, *wbp, *wbg, *wz, *wg;  // stored transposed [N,K]
};

static void run_trimul(float* zcur, const float* maskp, const MulW& w, const WS& ws,
                       int outgoing, hipStream_t st) {
  int srcT = outgoing ? 0 : 1;
  layernorm_kernel<<<NN / 8, 256, 0, st>>>(zcur, w.ln_in_g, w.ln_in_b, ws.H0);
  // a = mask * (zl@wap+bap) * sigmoid(zl@wag+bag), repacked bf16 [c,i,k]
  gemm(st, ws.H0, 0, CT, 0, 0, w.wap, 0, CT, 0, 0, 1, ws.F0, nullptr, 0, CT, 0, 0, 0,
       w.bap, nullptr, 0, 0, 1.f, NN, CT, CT, 1, 1);
  gemm(st, ws.H0, 0, CT, 0, 0, w.wag, 0, CT, 0, 0, 1, ws.F1, nullptr, 0, CT, 0, 0, 0,
       w.bag, nullptr, 0, 2, 1.f, NN, CT, CT, 1, 1);
  combine_pack_t_kernel<<<4096, 256, 0, st>>>(ws.F0, ws.F1, maskp, ws.H1, srcT);
  // b likewise
  gemm(st, ws.H0, 0, CT, 0, 0, w.wbp, 0, CT, 0, 0, 1, ws.F0, nullptr, 0, CT, 0, 0, 0,
       w.bbp, nullptr, 0, 0, 1.f, NN, CT, CT, 1, 1);
  gemm(st, ws.H0, 0, CT, 0, 0, w.wbg, 0, CT, 0, 0, 1, ws.F1, nullptr, 0, CT, 0, 0, 0,
       w.bbg, nullptr, 0, 2, 1.f, NN, CT, CT, 1, 1);
  combine_pack_t_kernel<<<4096, 256, 0, st>>>(ws.F0, ws.F1, maskp, ws.H2, srcT);
  // x[c,i,j] = A_c @ B_c^T : 128 batched 256x256x256 NT GEMMs
  gemm(st, ws.H1, 0, NRES, (long)NRES * NRES, 0,
           ws.H2, 0, NRES, (long)NRES * NRES, 0, 1,
           ws.F2, nullptr, 0, NRES, (long)NRES * NRES, 0, 0,
           nullptr, nullptr, 0, 0, 1.f, NRES, NRES, NRES, 1, CT);
  ln_perm_kernel<<<NN / 8, 256, 0, st>>>(ws.F2, w.ln_out_g, w.ln_out_b, ws.H3);
  gemm(st, ws.H3, 0, CT, 0, 0, w.wz, 0, CT, 0, 0, 1, ws.F0, nullptr, 0, CT, 0, 0, 0,
       w.bz, nullptr, 0, 0, 1.f, NN, CT, CT, 1, 1);
  gemm(st, ws.H0, 0, CT, 0, 0, w.wg, 0, CT, 0, 0, 1, ws.F1, nullptr, 0, CT, 0, 0, 0,
       w.bg, nullptr, 0, 2, 1.f, NN, CT, CT, 1, 1);
  mul_add_residual_kernel<<<4096, 256, 0, st>>>(zcur, ws.F0, ws.F1, (long)NN * CT);
}

extern "C" void kernel_launch(void* const* d_in, const int* in_sizes, int n_in,
                              void* d_out, int out_size, void* d_ws, size_t ws_size,
                              hipStream_t stream) {
  (void)in_sizes; (void)n_in; (void)out_size; (void)ws_size;

  // input order: z, mask, att_start{ln_g,ln_b,wb,wq,wk,wv,wg,bg,wo,bo},
  // att_end{...}, mul_out{ln_in_g,ln_in_b,wap,bap,wag,bag,wbp,bbp,wbg,bbg,
  // ln_out_g,ln_out_b,wz,bz,wg,bg}, mul_in{...}, trans{ln_g,ln_b,w1,b1,w2,b2}
  const float* mask = (const float*)d_in[1];
  float* zcur = (float*)d_out;
  auto F = [&](int i) { return (const float*)d_in[i]; };

  // ---- workspace arena ----
  char*  wsb = (char*)d_ws;
  size_t off = 0;
  auto alloc = [&](size_t bytes) -> char* {
    char* p = wsb + off;
    off += (bytes + 255) & ~(size_t)255;
    return p;
  };
  __bf16* warena = (__bf16*)alloc(1u << 20);
  WS ws;
  ws.tb    = (float*)alloc((size_t)NHEADS * NN * 4);
  ws.maskT = (float*)alloc((size_t)NN * 4);
  ws.H0 = (__bf16*)alloc((size_t)NN * CT * 2);
  ws.H1 = (__bf16*)alloc((size_t)NN * CT * 2);
  ws.H2 = (__bf16*)alloc((size_t)NN * CT * 2);
  ws.H3 = (__bf16*)alloc((size_t)NN * CT * 2);
  ws.H4 = (__bf16*)alloc((size_t)NN * CT * 2);
  ws.HA = (__bf16*)alloc((size_t)CHUNK * NHEADS * NRES * NRES * 2);
  ws.F0 = (float*)alloc((size_t)NN * CT * 4);
  ws.F1 = (float*)alloc((size_t)NN * CT * 4);
  ws.F2 = (float*)alloc((size_t)NN * CT * 4);
  ws.F3 = (float*)alloc((size_t)NN * CT * 4);  // also scores chunk (CHUNK*H*256*256 f32)

  // ---- convert GEMM weights to bf16 (transposed) once per call ----
  size_t wpos = 0;
  auto cvt_t = [&](int idx, int K, int N) -> __bf16* {
    __bf16* p = warena + wpos;
    long n = (long)K * N;
    wpos += (size_t)n;
    f32_to_bf16_t_kernel<<<dim3((unsigned)((n + 255) / 256)), 256, 0, stream>>>(F(idx), p, K, N);
    return p;
  };

  AttW aS = { F(2),  F(3),  F(4),  F(9),  F(11),
              cvt_t(5, CT, CT), cvt_t(6, CT, CT), cvt_t(7, CT, CT), cvt_t(8, CT, CT),
              cvt_t(10, CT, CT) };
  AttW aE = { F(12), F(13), F(14), F(19), F(21),
              cvt_t(15, CT, CT), cvt_t(16, CT, CT), cvt_t(17, CT, CT), cvt_t(18, CT, CT),
              cvt_t(20, CT, CT) };
  MulW mO = { F(22), F(23), F(25), F(27), F(29), F(31), F(32), F(33), F(35), F(37),
              cvt_t(24, CT, CT), cvt_t(26, CT, CT), cvt_t(28, CT, CT), cvt_t(30, CT, CT),
              cvt_t(34, CT, CT), cvt_t(36, CT, CT) };
  MulW mI = { F(38), F(39), F(41), F(43), F(45), F(47), F(48), F(49), F(51), F(53),
              cvt_t(40, CT, CT), cvt_t(42, CT, CT), cvt_t(44, CT, CT), cvt_t(46, CT, CT),
              cvt_t(50, CT, CT), cvt_t(52, CT, CT) };
  const float *tr_ln_g = F(54), *tr_ln_b = F(55), *tr_b1 = F(57), *tr_b2 = F(59);
  __bf16* tr_w1 = cvt_t(56, CT, 4 * CT);   // [512,128] bf16
  __bf16* tr_w2 = cvt_t(58, 4 * CT, CT);   // [128,512] bf16

  // residual state lives in d_out
  (void)hipMemcpyAsync(zcur, d_in[0], (size_t)NN * CT * sizeof(float),
                       hipMemcpyDeviceToDevice, stream);

  // 1) triangle attention, starting node: residual-add fused into wo epilogue
  run_attention(zcur, mask, aS, ws, zcur, 1, stream);

  // 2) triangle attention, ending node: run on transposed pair rep
  transpose_pair_kernel<<<4096, 256, 0, stream>>>(zcur, ws.F1);
  transpose_mask_kernel<<<NN / 256, 256, 0, stream>>>(mask, ws.maskT);
  run_attention(ws.F1, ws.maskT, aE, ws, ws.F0, 0, stream);
  transpose_add_pair_kernel<<<4096, 256, 0, stream>>>(ws.F0, zcur);

  // 3/4) triangle multiplication, outgoing then incoming
  run_trimul(zcur, mask, mO, ws, 1, stream);
  run_trimul(zcur, mask, mI, ws, 0, stream);

  // 5) transition (128 -> 512 relu -> 128), chunked so hidden fits a 16MB slab
  layernorm_kernel<<<NN / 8, 256, 0, stream>>>(zcur, tr_ln_g, tr_ln_b, ws.H0);
  const int RCH = 16384;
  for (int r0 = 0; r0 < NN; r0 += RCH) {
    gemm(stream, ws.H0, (long)r0 * CT, CT, 0, 0, tr_w1, 0, CT, 0, 0, 1,
         nullptr, ws.H1, 0, 4 * CT, 0, 0, 0, tr_b1, nullptr, 0, 1, 1.f,
         RCH, 4 * CT, CT, 1, 1);
    gemm(stream, ws.H1, 0, 4 * CT, 0, 0, tr_w2, 0, 4 * CT, 0, 0, 1,
         zcur, nullptr, (long)r0 * CT, CT, 0, 0, 1, tr_b2, mask, r0, 0, 1.f,
         RCH, CT, 4 * CT, 1, 1);
  }
}
